// PINet_39376260169811
// MI455X (gfx1250) — compile-verified, matrix operand order checked by reading, not run on hardware
//
#include <hip/hip_runtime.h>

// ---------------------------------------------------------------------------
// MI455X (gfx1250, wave32) PINet forward.
// Fused PI-net (31->256->256->256 -> key/val): a 64x256 activation tile lives
// in LDS and is transformed IN PLACE across all 5 layers; weight panels are
// streamed into double-buffered LDS by the Tensor Data Mover (TDM) with
// hardware padding (bank-conflict-free stride 72), overlapped with WMMA
// compute via TENSORcnt. All GEMMs: v_wmma_f32_16x16x32_f16 (fp32 acc).
// Per-head key/val projections folded into one GEMM via block-diagonal
// 256x256 weights. Attention (M=1 per head) is VALU + LDS.
// ---------------------------------------------------------------------------

typedef _Float16 half_t;
typedef __attribute__((ext_vector_type(16))) _Float16     v16h;
typedef __attribute__((ext_vector_type(8)))  _Float16     v8h;
typedef __attribute__((ext_vector_type(8)))  float        v8f;
typedef __attribute__((ext_vector_type(4)))  unsigned int v4u;
typedef __attribute__((ext_vector_type(8)))  int          v8i;
typedef __attribute__((ext_vector_type(4)))  int          v4i;

#define BATCH   4096
#define OBS_W   2304                  // 2048 objs + 256 others
#define ROWS_PI (BATCH * 64)          // 262144

#define WSTR 72    // weight panel row stride (halves): 64 + 8 pad (bank spread)
#define TSTR 264   // activation tile row stride (halves): 256 + 8 pad

__device__ __forceinline__ v16h load_frag16(const half_t* p) {
    v8h lo = *(const v8h*)(p);
    v8h hi = *(const v8h*)(p + 16);
    v16h r;
#pragma unroll
    for (int i = 0; i < 8; ++i) { r[i] = lo[i]; r[8 + i] = hi[i]; }
    return r;
}

// ---------------------------------------------------------------------------
// TDM: DMA one 256-row x 64-half panel of W^T (row stride 256 halves) from
// global into LDS, padding 16B after every 128B row -> LDS stride WSTR=72.
// D# packing per CDNA5 ISA ch.8 (group0: count/lds/global/type, group1:
// sizes/strides/pad, groups 2-3 zero for 2D). Issued by ONE wave only.
// ---------------------------------------------------------------------------
__device__ __forceinline__ void tdm_load_panel(const half_t* Wg, int k0,
                                               half_t* sWdst)
{
    const unsigned long long ga =
        (unsigned long long)(size_t)(Wg + k0);          // tile start (bytes)
    const unsigned int lds = (unsigned int)(size_t)sWdst; // low 32b = LDS addr

    v4u g0;
    g0.x = 1u;                                          // count=1, user D#
    g0.y = lds;                                         // lds_addr
    g0.z = (unsigned int)(ga & 0xffffffffull);          // global_addr[31:0]
    g0.w = (unsigned int)((ga >> 32) & 0x01ffffffull)   // global_addr[56:32]
         | 0x80000000u;                                 // type=2 ("image")

    v8i g1;
    g1[0] = (3 << 16)      // data_size = 8B units
          | (1 << 20)      // pad_enable
          | (4 << 22)      // pad_interval: every 32 DWORDs (128B row)
          | (3 << 25);     // pad_amount: 4 DWORDs (16B) -> stride 144B = WSTR
    g1[1] = (16 << 16);    // tensor_dim0 = 16 units (bits 79:48 low half)
    g1[2] = (256 << 16);   // tensor_dim1 = 256 rows
    g1[3] = (16 << 16);    // tile_dim0 = 16 units (128B)
    g1[4] = 256;           // tile_dim1 = 256 rows, tile_dim2 = 0
    g1[5] = 64;            // tensor_dim0_stride = 64 units (512B = 256 halves)
    g1[6] = 0;
    g1[7] = 0;

    v4i gz = {0, 0, 0, 0};
#if defined(__clang_major__) && (__clang_major__ >= 23)
    v8i gz8 = {0, 0, 0, 0, 0, 0, 0, 0};
    __builtin_amdgcn_tensor_load_to_lds(g0, g1, gz, gz, gz8, 0);
#else
    __builtin_amdgcn_tensor_load_to_lds(g0, g1, gz, gz, 0);
#endif
}

// ---------------------------------------------------------------------------
// One K=256 layer on the in-place 64x256 LDS tile. Weight panels double-
// buffered via TDM; panel p+1 DMAs while all 8 waves compute on panel p.
// ---------------------------------------------------------------------------
__device__ __forceinline__ void layer256_tdm(const half_t* __restrict__ Wg,
                                             const half_t* sT,
                                             half_t* sW0, half_t* sW1,
                                             v8f acc[8], int t, int wm, int wn,
                                             int l15, int kh)
{
    const v8f vz = {0.f, 0.f, 0.f, 0.f, 0.f, 0.f, 0.f, 0.f};
#pragma unroll
    for (int j = 0; j < 8; ++j) acc[j] = vz;

    const bool issuer = (t < 32);     // wave 0 (uniform branch; TDM ignores EXEC)

    __syncthreads();                  // prior readers of sW0/sW1 are done
    if (issuer) tdm_load_panel(Wg, 0, sW0);

#pragma unroll
    for (int p = 0; p < 4; ++p) {
        half_t* sWc = (p & 1) ? sW1 : sW0;
        half_t* sWn = (p & 1) ? sW0 : sW1;
        if (p + 1 < 4) {
            if (issuer) tdm_load_panel(Wg, (p + 1) * 64, sWn);
            if (issuer) __builtin_amdgcn_s_wait_tensorcnt((short)1);
        } else {
            if (issuer) __builtin_amdgcn_s_wait_tensorcnt((short)0);
        }
        __syncthreads();              // panel p visible to all waves
#pragma unroll
        for (int ks = 0; ks < 64; ks += 32) {
            v16h a = load_frag16(sT + (wm * 16 + l15) * TSTR + p * 64 + ks + kh);
#pragma unroll
            for (int j = 0; j < 8; ++j) {
                v16h b = load_frag16(&sWc[(wn + j * 16 + l15) * WSTR + ks + kh]);
                acc[j] = __builtin_amdgcn_wmma_f32_16x16x32_f16(
                    false, a, false, b, (short)0, acc[j], false, false);
            }
        }
        __syncthreads();              // sWc reusable at p+2; tile reads done (p=3)
    }
    // trailing barrier above also licenses IN-PLACE epilogue writes to sT
}

// Epilogue: +bias, optional relu, optional per-global-row scale; store to the
// LDS tile (in place) and/or to global f16 (stride 256). Pointer null-ness is
// compile-time constant at every call site.
__device__ __forceinline__ void epilogue8(const v8f acc[8],
                                          const float* __restrict__ bias,
                                          const float* __restrict__ rowscale,
                                          int do_relu, half_t* sOut,
                                          half_t* __restrict__ gOut,
                                          int rowBlk, int wm, int wn,
                                          int l15, int hi8)
{
#pragma unroll
    for (int j = 0; j < 8; ++j) {
        const int n = wn + j * 16 + l15;
        const float bsv = bias[n];
#pragma unroll
        for (int v = 0; v < 8; ++v) {
            const int lrow = wm * 16 + hi8 + v;
            float x = acc[j][v] + bsv;
            if (do_relu)  x = fmaxf(x, 0.f);
            if (rowscale) x *= rowscale[rowBlk + lrow];
            if (sOut) sOut[lrow * TSTR + n] = (half_t)x;
            if (gOut) gOut[(size_t)(rowBlk + lrow) * 256 + n] = (half_t)x;
        }
    }
}

// ---------------------------------------------------------------------------
// Fused PI-net: per block, 64 object rows through all 5 layers.
// ---------------------------------------------------------------------------
__global__ __launch_bounds__(256)
void pinet_fused(const half_t* __restrict__ A0,
                 const half_t* __restrict__ pW1T, const half_t* __restrict__ pW2T,
                 const half_t* __restrict__ pW3T, const half_t* __restrict__ WkT,
                 const half_t* __restrict__ WvT,
                 const float* __restrict__ pb1, const float* __restrict__ pb2,
                 const float* __restrict__ pb3, const float* __restrict__ bk,
                 const float* __restrict__ bv,  const float* __restrict__ mask,
                 half_t* __restrict__ keyOut, half_t* __restrict__ valOut)
{
    __shared__ __align__(16) half_t sW0[256 * WSTR];   // 36 KB panel (TDM dst)
    __shared__ __align__(16) half_t sW1[256 * WSTR];   // 36 KB panel (TDM dst)
    __shared__ __align__(16) half_t sT [64 * TSTR];    // 33.8 KB in-place tile

    const int t    = threadIdx.x;
    const int lane = t & 31;
    const int wid  = t >> 5;
    const int wm   = wid >> 1;          // 0..3 -> 16-row strip
    const int wn   = (wid & 1) << 7;    // 0 or 128 -> 128-col strip
    const int l15  = lane & 15;
    const int kh   = (lane & 16) ? 8 : 0;   // ISA f16 A/B lane K-octet select
    const int hi8  = kh;                    // C/D row offset for lanes 16..31
    const int rowBlk = blockIdx.x * 64;

    v8f acc[8];
    const v8f vz = {0.f, 0.f, 0.f, 0.f, 0.f, 0.f, 0.f, 0.f};

    // ---- layer 1: h1 = relu(A0 @ pW1 + pb1), K = 32: cooperative copy ----
    for (int i = t; i < 1024; i += 256) {              // 256 n-rows x 4 uint4
        const int n  = i >> 2;
        const int kk = (i & 3) << 3;
        *(uint4*)&sW0[n * WSTR + kk] = *(const uint4*)&pW1T[n * 32 + kk];
    }
    __syncthreads();
#pragma unroll
    for (int j = 0; j < 8; ++j) acc[j] = vz;
    {
        v16h a = load_frag16(A0 + (size_t)(rowBlk + wm * 16 + l15) * 32 + kh);
#pragma unroll
        for (int j = 0; j < 8; ++j) {
            v16h b = load_frag16(&sW0[(wn + j * 16 + l15) * WSTR + kh]);
            acc[j] = __builtin_amdgcn_wmma_f32_16x16x32_f16(
                false, a, false, b, (short)0, acc[j], false, false);
        }
    }
    epilogue8(acc, pb1, nullptr, 1, sT, nullptr, rowBlk, wm, wn, l15, hi8);

    // ---- layer 2: h2 = relu(h1 @ pW2 + pb2)  (in-place on sT) ----
    layer256_tdm(pW2T, sT, sW0, sW1, acc, t, wm, wn, l15, kh);
    epilogue8(acc, pb2, nullptr, 1, sT, nullptr, rowBlk, wm, wn, l15, hi8);

    // ---- layer 3: emb = (h2 @ pW3 + pb3) * mask  (in-place) ----
    layer256_tdm(pW3T, sT, sW0, sW1, acc, t, wm, wn, l15, kh);
    epilogue8(acc, pb3, mask, 0, sT, nullptr, rowBlk, wm, wn, l15, hi8);

    // ---- layer 4: key = emb @ blockdiag(Wk) + bk -> global ----
    layer256_tdm(WkT, sT, sW0, sW1, acc, t, wm, wn, l15, kh);
    epilogue8(acc, bk, nullptr, 0, nullptr, keyOut, rowBlk, wm, wn, l15, hi8);

    // ---- layer 5: val = (emb @ blockdiag(Wv) + bv) * mask -> global ----
    layer256_tdm(WvT, sT, sW0, sW1, acc, t, wm, wn, l15, kh);
    epilogue8(acc, bv, mask, 0, nullptr, valOut, rowBlk, wm, wn, l15, hi8);
}

// ---------------------------------------------------------------------------
// Weight prep: f32 -> f16 transposed W^T[n][k]; pW1 padded to K=32 with a
// zero row; Wk/Wv expanded to block-diagonal 256x256 (transposed).
// ---------------------------------------------------------------------------
__global__ __launch_bounds__(256)
void prep_weights(const float* __restrict__ oW1, const float* __restrict__ oW2,
                  const float* __restrict__ pW1, const float* __restrict__ pW2,
                  const float* __restrict__ pW3, const float* __restrict__ Wq,
                  const float* __restrict__ Wk,  const float* __restrict__ Wv,
                  half_t* oW1T, half_t* oW2T, half_t* pW1T, half_t* pW2T,
                  half_t* pW3T, half_t* WqT, half_t* WkT, half_t* WvT)
{
    const int t = threadIdx.x;
    for (int i = t; i < 256 * 256; i += 256) {
        WkT[i] = (half_t)0.0f;
        WvT[i] = (half_t)0.0f;
    }
    for (int i = t; i < 256 * 256; i += 256) {
        const int k = i >> 8, n = i & 255;
        oW1T[n * 256 + k] = (half_t)oW1[i];
        oW2T[n * 256 + k] = (half_t)oW2[i];
        pW2T[n * 256 + k] = (half_t)pW2[i];
        pW3T[n * 256 + k] = (half_t)pW3[i];
        WqT [n * 256 + k] = (half_t)Wq [i];
    }
    for (int i = t; i < 31 * 256; i += 256) {
        const int k = i >> 8, n = i & 255;
        pW1T[n * 32 + k] = (half_t)pW1[i];
    }
    pW1T[t * 32 + 31] = (half_t)0.0f;
    __syncthreads();
    for (int i = t; i < 4 * 64 * 64; i += 256) {
        const int h = i >> 12;
        const int r = i & 4095;
        const int d = r >> 6;
        const int e = r & 63;
        const size_t dst = (size_t)(h * 64 + e) * 256 + h * 64 + d;
        WkT[dst] = (half_t)Wk[i];
        WvT[dst] = (half_t)Wv[i];
    }
}

// ---------------------------------------------------------------------------
// Gather/convert: obs f32 -> A0 f16 (262144x32; col 31 = mask, killed by the
// zeroed pW1 K-row), others f16 (4096x256), mask f32 (262144).
// ---------------------------------------------------------------------------
__global__ __launch_bounds__(256)
void gather_convert(const float* __restrict__ obs,
                    half_t* __restrict__ A0, half_t* __restrict__ others16,
                    float* __restrict__ mask)
{
    const int b = blockIdx.x;
    const int t = threadIdx.x;
    const float* row = obs + (size_t)b * OBS_W;
    for (int i = t; i < 2048; i += 256)
        A0[(size_t)b * 2048 + i] = (half_t)row[i];
    others16[(size_t)b * 256 + t] = (half_t)row[2048 + t];
    if (t < 64)
        mask[b * 64 + t] = row[t * 32 + 31];
}

// ---------------------------------------------------------------------------
// Generic staged WMMA GEMM (N=256, lda=K) for the small others-MLP path.
// ---------------------------------------------------------------------------
__global__ __launch_bounds__(256, 2)
void wmma_gemm_n256(const half_t* __restrict__ A, int K,
                    const half_t* __restrict__ WT,
                    const float* __restrict__ bias,
                    int do_relu,
                    half_t* __restrict__ out16,
                    float* __restrict__ out32, int ldc32)
{
    __shared__ __align__(16) half_t sW[256 * WSTR];

    const int t    = threadIdx.x;
    const int lane = t & 31;
    const int wid  = t >> 5;
    const int wm   = wid >> 2;
    const int wn   = (wid & 3) << 6;
    const int row0 = blockIdx.x * 32 + wm * 16;
    const int l15  = lane & 15;
    const int kh   = (lane & 16) ? 8 : 0;
    const int hi8  = kh;

    const v8f vz = {0.f, 0.f, 0.f, 0.f, 0.f, 0.f, 0.f, 0.f};
    v8f acc[4];
#pragma unroll
    for (int j = 0; j < 4; ++j) acc[j] = vz;

    const size_t arow = (size_t)(row0 + l15) * (size_t)K;

    for (int k0 = 0; k0 < K; k0 += 64) {
        int kc = K - k0; if (kc > 64) kc = 64;
        __syncthreads();
        const int vecs_per_n = kc >> 3;
        for (int i = t; i < 256 * vecs_per_n; i += 256) {
            const int n  = i / vecs_per_n;
            const int kk = (i - n * vecs_per_n) << 3;
            *(uint4*)&sW[n * WSTR + kk] = *(const uint4*)&WT[(size_t)n * K + k0 + kk];
        }
        __syncthreads();
#pragma unroll
        for (int ks = 0; ks < 64; ks += 32) {
            if (ks >= kc) break;
            v16h a = load_frag16(A + arow + (k0 + ks + kh));
#pragma unroll
            for (int j = 0; j < 4; ++j) {
                v16h b = load_frag16(&sW[(wn + j * 16 + l15) * WSTR + ks + kh]);
                acc[j] = __builtin_amdgcn_wmma_f32_16x16x32_f16(
                    false, a, false, b, (short)0, acc[j], false, false);
            }
        }
    }

#pragma unroll
    for (int j = 0; j < 4; ++j) {
        const int n = wn + j * 16 + l15;
        const float bsv = bias ? bias[n] : 0.0f;
#pragma unroll
        for (int v = 0; v < 8; ++v) {
            const int row = row0 + hi8 + v;
            float x = acc[j][v] + bsv;
            if (do_relu) x = fmaxf(x, 0.0f);
            if (out16) out16[(size_t)row * 256 + n] = (half_t)x;
            if (out32) out32[(size_t)row * ldc32 + n] = x;
        }
    }
}

// ---------------------------------------------------------------------------
// Attention: one block per batch element; logits -> masked softmax -> weighted
// sum, all in LDS (q is 1x64 per head: WMMA-hostile, VALU-cheap).
// ---------------------------------------------------------------------------
__global__ __launch_bounds__(256)
void attention_kernel(const half_t* __restrict__ q16,
                      const half_t* __restrict__ key16,
                      const half_t* __restrict__ val16,
                      const float*  __restrict__ mask,
                      float* __restrict__ out)
{
    __shared__ float sLog[4 * 64];
    __shared__ float sExp[4 * 64];

    const int b = blockIdx.x;
    const int t = threadIdx.x;
    const int h = t >> 6;
    const int x = t & 63;

    const half_t* qp = q16   + (size_t)b * 256 + h * 64;
    const half_t* kp = key16 + ((size_t)b * 64 + x) * 256 + h * 64;
    float acc = 0.0f;
#pragma unroll 8
    for (int d = 0; d < 64; ++d)
        acc += (float)qp[d] * (float)kp[d];
    const float mv = mask[b * 64 + x];
    sLog[t] = (mv == 0.0f) ? -1.0e9f : acc * 0.0625f;   // 1/sqrt(256)
    __syncthreads();

    float mx = -3.4e38f;
#pragma unroll 8
    for (int i = 0; i < 64; ++i) mx = fmaxf(mx, sLog[h * 64 + i]);
    sExp[t] = expf(sLog[t] - mx);
    __syncthreads();

    float sum = 0.0f;
#pragma unroll 8
    for (int i = 0; i < 64; ++i) sum += sExp[h * 64 + i];

    const half_t* vp = val16 + (size_t)b * 64 * 256 + h * 64 + x;
    float o = 0.0f;
#pragma unroll 8
    for (int n = 0; n < 64; ++n)
        o += sExp[h * 64 + n] * (float)vp[(size_t)n * 256];

    out[(size_t)b * 512 + 256 + h * 64 + x] = o / sum;
}

// ---------------------------------------------------------------------------
// Host-side orchestration
// ---------------------------------------------------------------------------
extern "C" void kernel_launch(void* const* d_in, const int* in_sizes, int n_in,
                              void* d_out, int out_size, void* d_ws, size_t ws_size,
                              hipStream_t stream) {
    const float* obs = (const float*)d_in[0];
    const float* oW1 = (const float*)d_in[1];
    const float* ob1 = (const float*)d_in[2];
    const float* oW2 = (const float*)d_in[3];
    const float* ob2 = (const float*)d_in[4];
    const float* pW1 = (const float*)d_in[5];
    const float* pb1 = (const float*)d_in[6];
    const float* pW2 = (const float*)d_in[7];
    const float* pb2 = (const float*)d_in[8];
    const float* pW3 = (const float*)d_in[9];
    const float* pb3 = (const float*)d_in[10];
    const float* Wq  = (const float*)d_in[11];
    const float* bq  = (const float*)d_in[12];
    const float* Wk  = (const float*)d_in[13];
    const float* bk  = (const float*)d_in[14];  // (H,HD) flat == tiled 256 bias
    const float* Wv  = (const float*)d_in[15];
    const float* bv  = (const float*)d_in[16];
    float* out = (float*)d_out;

    char* w = (char*)d_ws;
    auto alloc = [&](size_t bytes) -> void* {
        void* p = (void*)w;
        w += (bytes + 255) & ~(size_t)255;
        return p;
    };
    half_t* oW1T = (half_t*)alloc(256 * 256 * 2);
    half_t* oW2T = (half_t*)alloc(256 * 256 * 2);
    half_t* pW1T = (half_t*)alloc(256 * 32  * 2);
    half_t* pW2T = (half_t*)alloc(256 * 256 * 2);
    half_t* pW3T = (half_t*)alloc(256 * 256 * 2);
    half_t* WqT  = (half_t*)alloc(256 * 256 * 2);
    half_t* WkT  = (half_t*)alloc(256 * 256 * 2);
    half_t* WvT  = (half_t*)alloc(256 * 256 * 2);
    half_t* A0   = (half_t*)alloc((size_t)ROWS_PI * 32 * 2);   // 16.8 MB
    half_t* oth  = (half_t*)alloc((size_t)BATCH * 256 * 2);
    float*  mask = (float*) alloc((size_t)ROWS_PI * 4);
    half_t* keyB = (half_t*)alloc((size_t)ROWS_PI * 256 * 2);  // 128 MB
    half_t* valB = (half_t*)alloc((size_t)ROWS_PI * 256 * 2);  // 128 MB
    half_t* o1   = (half_t*)alloc((size_t)BATCH * 256 * 2);
    half_t* oe16 = (half_t*)alloc((size_t)BATCH * 256 * 2);
    half_t* q16  = (half_t*)alloc((size_t)BATCH * 256 * 2);

    dim3 blk(256);

    prep_weights<<<1, blk, 0, stream>>>(oW1, oW2, pW1, pW2, pW3, Wq, Wk, Wv,
                                        oW1T, oW2T, pW1T, pW2T, pW3T, WqT, WkT, WvT);
    gather_convert<<<BATCH, blk, 0, stream>>>(obs, A0, oth, mask);

    // Fused PI-net: A0 -> key/val, 64 rows/block, TDM-streamed weights
    pinet_fused<<<ROWS_PI / 64, blk, 0, stream>>>(
        A0, pW1T, pW2T, pW3T, WkT, WvT, pb1, pb2, pb3, bk, bv, mask, keyB, valB);

    // Others MLP + q (small path)
    const int gOTH = BATCH / 32;
    wmma_gemm_n256<<<gOTH, blk, 0, stream>>>(oth, 256, oW1T, ob1, 1,
                                             o1, nullptr, 0);
    wmma_gemm_n256<<<gOTH, blk, 0, stream>>>(o1, 256, oW2T, ob2, 0,
                                             oe16, out, 512);
    wmma_gemm_n256<<<gOTH, blk, 0, stream>>>(oe16, 256, WqT, bq, 0,
                                             q16, nullptr, 0);

    // attention -> d_out[:, 256:512]
    attention_kernel<<<BATCH, blk, 0, stream>>>(q16, keyB, valB, mask, out);
}
